// SparseViT_64622077935663
// MI455X (gfx1250) — matrix operand order repaired and checked by reading, not compiled, not run on hardware
//
#include <hip/hip_runtime.h>
#include <hip/hip_bf16.h>
#include <math.h>

typedef __attribute__((ext_vector_type(16))) _Float16 v16h;
typedef __attribute__((ext_vector_type(8)))  _Float16 v8h;
typedef __attribute__((ext_vector_type(4)))  _Float16 h4;
typedef __attribute__((ext_vector_type(8)))  float    v8f;
typedef __attribute__((ext_vector_type(4)))  float    f32x4;

#define LDSW 40   // LDS row stride in halves (32 + 8 pad, keeps 16B alignment)

// ---------------------------------------------------------------------------
// Generic batched GEMM:  C = alpha * A @ B' (+bias) (gelu?) (+resid)
// A: f32 [M,K] row-major (lda), per-batch/head strides sAb/sAh
// B: f32 [K,N] row-major (ldb) or, if transB, element (k,n) = B[n*ldb + k]
// C: f32 [M,N] (ldc). grid.z = nb*Hh, z -> (b = z/Hh, h = z%Hh)
// ---------------------------------------------------------------------------
struct GemmP {
  const float* A; const float* B; float* C;
  const float* bias; const float* resid;
  long long lda, ldb, ldc, ldr;
  long long sAb, sAh, sBb, sBh, sCb, sCh, sRb, sRh;
  int M, N, K, Hh, transB, gelu;
  float alpha;
};

__global__ __launch_bounds__(256) void gemm_wmma_kernel(GemmP p) {
  __shared__ __align__(16) _Float16 As[128 * LDSW];
  __shared__ __align__(16) _Float16 Bs[128 * LDSW];

  const int t = threadIdx.x;
  const int bz = blockIdx.z;
  const int bb = bz / p.Hh;
  const int hh = bz % p.Hh;

  const float* A = p.A + (long long)bb * p.sAb + (long long)hh * p.sAh;
  const float* B = p.B + (long long)bb * p.sBb + (long long)hh * p.sBh;
  float*       C = p.C + (long long)bb * p.sCb + (long long)hh * p.sCh;
  const float* R = p.resid ? (p.resid + (long long)bb * p.sRb + (long long)hh * p.sRh) : nullptr;

  const int m0 = blockIdx.y * 128;
  const int n0 = blockIdx.x * 128;

  const int wave = t >> 5, lane = t & 31;
  const int wm = wave & 1, wn = wave >> 1;       // 2 waves in M, 4 in N
  const int mBase = wm * 64, nBase = wn * 32;    // 64x32 per wave = 4x2 WMMA tiles

  // uniform fast-path predicates (independent of k0 except K bound)
  const bool aVec  = (((p.lda & 3) == 0) && ((((unsigned long long)A) & 15) == 0) && (m0 + 128 <= p.M));
  const bool bVecT = (((p.ldb & 3) == 0) && ((((unsigned long long)B) & 15) == 0) && (n0 + 128 <= p.N));
  const bool bFull = (n0 + 128 <= p.N);

  v8f zero;
  #pragma unroll
  for (int q = 0; q < 8; q++) zero[q] = 0.0f;
  v8f acc[4][2];
  #pragma unroll
  for (int i = 0; i < 4; i++)
    #pragma unroll
    for (int j = 0; j < 2; j++) acc[i][j] = zero;

  for (int k0 = 0; k0 < p.K; k0 += 32) {
    const bool kFull = (k0 + 32 <= p.K);

    // ---- stage A tile (128 x 32), f32 -> f16 ----
    if (aVec && kFull) {
      #pragma unroll
      for (int i = 0; i < 4; i++) {
        int e = i * 256 + t;                 // 1024 float4 chunks
        int r = e >> 3, c4 = (e & 7) * 4;
        f32x4 v = *(const f32x4*)(A + (long long)(m0 + r) * p.lda + (k0 + c4));
        h4 hv;
        #pragma unroll
        for (int j = 0; j < 4; j++) hv[j] = (_Float16)v[j];
        *(h4*)&As[r * LDSW + c4] = hv;
      }
    } else {
      #pragma unroll
      for (int i = 0; i < 16; i++) {
        int e = i * 256 + t;
        int r = e >> 5, kk = e & 31;
        int gm = m0 + r, gk = k0 + kk;
        bool ok = (gm < p.M) && (gk < p.K);
        long long off = ok ? ((long long)gm * p.lda + gk) : 0;
        float v = A[off];
        As[r * LDSW + kk] = ok ? (_Float16)v : (_Float16)0.0f;
      }
    }

    // ---- stage B tile (32 x 128), stored K-major per column: Bs[n*LDSW + k] ----
    if (!p.transB) {
      if (bFull && kFull) {
        #pragma unroll
        for (int i = 0; i < 4; i++) {
          int e = i * 256 + t;               // 1024 4-deep column chunks
          int n = e & 127, kk4 = (e >> 7) * 4;
          const float* bp = B + (long long)(k0 + kk4) * p.ldb + (n0 + n);
          h4 hv;
          #pragma unroll
          for (int j = 0; j < 4; j++) hv[j] = (_Float16)bp[(long long)j * p.ldb];
          *(h4*)&Bs[n * LDSW + kk4] = hv;
        }
      } else {
        #pragma unroll
        for (int i = 0; i < 16; i++) {
          int e = i * 256 + t;
          int kk = e >> 7, nn = e & 127;
          int gk = k0 + kk, gn = n0 + nn;
          bool ok = (gk < p.K) && (gn < p.N);
          long long off = ok ? ((long long)gk * p.ldb + gn) : 0;
          float v = B[off];
          Bs[nn * LDSW + kk] = ok ? (_Float16)v : (_Float16)0.0f;
        }
      }
    } else {
      if (bVecT && kFull) {
        #pragma unroll
        for (int i = 0; i < 4; i++) {
          int e = i * 256 + t;
          int n = e >> 3, c4 = (e & 7) * 4;
          f32x4 v = *(const f32x4*)(B + (long long)(n0 + n) * p.ldb + (k0 + c4));
          h4 hv;
          #pragma unroll
          for (int j = 0; j < 4; j++) hv[j] = (_Float16)v[j];
          *(h4*)&Bs[n * LDSW + c4] = hv;
        }
      } else {
        #pragma unroll
        for (int i = 0; i < 16; i++) {
          int e = i * 256 + t;
          int nn = e >> 5, kk = e & 31;
          int gk = k0 + kk, gn = n0 + nn;
          bool ok = (gk < p.K) && (gn < p.N);
          long long off = ok ? ((long long)gn * p.ldb + gk) : 0;
          float v = B[off];
          Bs[nn * LDSW + kk] = ok ? (_Float16)v : (_Float16)0.0f;
        }
      }
    }
    __syncthreads();

    // ---- B fragments (ISA 16-bit B 32x16: lanes 0-15 K 0..7,16..23, lanes 16-31 K 8..15,24..31) ----
    v16h bfrag[2];
    #pragma unroll
    for (int nt = 0; nt < 2; nt++) {
      int n = nBase + nt * 16 + (lane & 15);
      int kb2 = (lane >> 4) * 8;
      const v8h* q0 = (const v8h*)&Bs[n * LDSW + kb2];
      v8h lo = q0[0], hi = q0[2];
      v16h bf;
      #pragma unroll
      for (int j = 0; j < 8; j++) { bf[j] = lo[j]; bf[8 + j] = hi[j]; }
      bfrag[nt] = bf;
    }
    // ---- A fragments (ISA 16-bit A 16x32) + WMMA ----
    const int arow = lane & 15;
    const int akb  = (lane >> 4) * 8;
    #pragma unroll
    for (int mt = 0; mt < 4; mt++) {
      int r = mBase + mt * 16 + arow;
      const v8h* p0 = (const v8h*)&As[r * LDSW + akb];
      const v8h* p1 = (const v8h*)&As[r * LDSW + 16 + akb];
      v8h lo = *p0, hi = *p1;
      v16h af;
      #pragma unroll
      for (int j = 0; j < 8; j++) { af[j] = lo[j]; af[8 + j] = hi[j]; }
      #pragma unroll
      for (int nt = 0; nt < 2; nt++) {
        acc[mt][nt] = __builtin_amdgcn_wmma_f32_16x16x32_f16(
            false, af, false, bfrag[nt], (short)0, acc[mt][nt], false, false);
      }
    }
    __syncthreads();
  }

  // ---- epilogue: alpha -> +bias -> gelu -> +resid -> store ----
  const int colL = lane & 15;
  const int rowHi = (lane >> 4) * 8;
  const bool interior = (m0 + 128 <= p.M) && (n0 + 128 <= p.N);
  #pragma unroll
  for (int mt = 0; mt < 4; mt++)
    #pragma unroll
    for (int nt = 0; nt < 2; nt++)
      #pragma unroll
      for (int j = 0; j < 8; j++) {
        int gm = m0 + mBase + mt * 16 + rowHi + j;
        int gn = n0 + nBase + nt * 16 + colL;
        if (interior || (gm < p.M && gn < p.N)) {
          float v = acc[mt][nt][j] * p.alpha;
          if (p.bias) v += p.bias[gn];
          if (p.gelu) v = 0.5f * v * (1.0f + erff(v * 0.70710678118654752f));
          if (R) v += R[(long long)gm * p.ldr + gn];
          C[(long long)gm * p.ldc + gn] = v;
        }
      }
}

// ---------------------------------------------------------------------------
// Elementwise / row kernels
// ---------------------------------------------------------------------------
__global__ void im2col_kernel(const float* __restrict__ x, float* __restrict__ xp) {
  long long idx = (long long)blockIdx.x * 256 + threadIdx.x;
  if (idx >= 6272LL * 768) return;
  int k = (int)(idx % 768);
  long long bn = idx / 768;
  int n = (int)(bn % 196);
  int b = (int)(bn / 196);
  int c = k >> 8, pi = (k >> 4) & 15, pj = k & 15;
  int gi = n / 14, gj = n % 14;
  xp[idx] = x[(((long long)b * 3 + c) * 224 + (gi * 16 + pi)) * 224 + (gj * 16 + pj)];
}

__global__ void assemble_kernel(const float* __restrict__ patches,
                                const float* __restrict__ cls,
                                const float* __restrict__ pos,
                                float* __restrict__ X) {
  long long idx = (long long)blockIdx.x * 256 + threadIdx.x;
  if (idx >= 32LL * 197 * 768) return;
  int e = (int)(idx % 768);
  long long bs = idx / 768;
  int s = (int)(bs % 197);
  int b = (int)(bs / 197);
  float v = (s == 0) ? cls[e] : patches[((long long)b * 196 + (s - 1)) * 768 + e];
  X[idx] = v + pos[(long long)s * 768 + e];
}

__global__ void ln_kernel(const float* __restrict__ in, float* __restrict__ out,
                          const float* __restrict__ s, const float* __restrict__ b,
                          int D, long long inStride, long long outStride, float eps) {
  long long row = blockIdx.x;
  const float* xr = in + row * inStride;
  float* yr = out + row * outStride;
  __shared__ float red[256];
  __shared__ float sMean, sInv;
  float sum = 0.0f, sq = 0.0f;
  for (int i = threadIdx.x; i < D; i += 256) { float v = xr[i]; sum += v; sq += v * v; }
  red[threadIdx.x] = sum; __syncthreads();
  for (int o = 128; o > 0; o >>= 1) { if (threadIdx.x < o) red[threadIdx.x] += red[threadIdx.x + o]; __syncthreads(); }
  if (threadIdx.x == 0) sMean = red[0] / (float)D;
  __syncthreads();
  red[threadIdx.x] = sq; __syncthreads();
  for (int o = 128; o > 0; o >>= 1) { if (threadIdx.x < o) red[threadIdx.x] += red[threadIdx.x + o]; __syncthreads(); }
  if (threadIdx.x == 0) {
    float var = red[0] / (float)D - sMean * sMean;
    sInv = rsqrtf(var + eps);
  }
  __syncthreads();
  float mean = sMean, inv = sInv;
  for (int i = threadIdx.x; i < D; i += 256)
    yr[i] = (xr[i] - mean) * inv * s[i] + b[i];
}

__global__ void softmax_kernel(float* __restrict__ data, int cols) {
  long long row = blockIdx.x;
  float* xr = data + row * (long long)cols;
  __shared__ float red[256];
  __shared__ float sv;
  float mx = -1e30f;
  for (int i = threadIdx.x; i < cols; i += 256) mx = fmaxf(mx, xr[i]);
  red[threadIdx.x] = mx; __syncthreads();
  for (int o = 128; o > 0; o >>= 1) { if (threadIdx.x < o) red[threadIdx.x] = fmaxf(red[threadIdx.x], red[threadIdx.x + o]); __syncthreads(); }
  if (threadIdx.x == 0) sv = red[0];
  __syncthreads();
  float m = sv, s = 0.0f;
  for (int i = threadIdx.x; i < cols; i += 256) { float e = expf(xr[i] - m); xr[i] = e; s += e; }
  red[threadIdx.x] = s; __syncthreads();
  for (int o = 128; o > 0; o >>= 1) { if (threadIdx.x < o) red[threadIdx.x] += red[threadIdx.x + o]; __syncthreads(); }
  if (threadIdx.x == 0) sv = red[0];
  __syncthreads();
  float inv = 1.0f / sv;
  for (int i = threadIdx.x; i < cols; i += 256) xr[i] *= inv;
}

__global__ void selmask_kernel(const float* __restrict__ g, const float* __restrict__ w,
                               const float* __restrict__ b2,
                               float* __restrict__ masks, float* __restrict__ out_masks) {
  long long row = blockIdx.x;
  __shared__ float red[256];
  float sum = 0.0f;
  for (int i = threadIdx.x; i < 768; i += 256) sum += g[row * 768 + i] * w[i];
  red[threadIdx.x] = sum; __syncthreads();
  for (int o = 128; o > 0; o >>= 1) { if (threadIdx.x < o) red[threadIdx.x] += red[threadIdx.x + o]; __syncthreads(); }
  if (threadIdx.x == 0) {
    float sl = red[0] + b2[0];
    float m = 1.0f / (1.0f + expf(-sl));
    masks[row] = m;
    out_masks[row] = m;
  }
}

__global__ void applymask_kernel(float* __restrict__ X, const float* __restrict__ masks) {
  long long idx = (long long)blockIdx.x * 256 + threadIdx.x;
  if (idx >= 6272LL * 768) return;
  int e = (int)(idx % 768);
  long long bn = idx / 768;
  int n = (int)(bn % 196);
  int b = (int)(bn / 196);
  X[((long long)b * 197 + 1 + n) * 768 + e] *= masks[bn];
}

// ---------------------------------------------------------------------------
// Host side
// ---------------------------------------------------------------------------
static inline void launch_gemm(hipStream_t st,
    const float* A, long long lda, long long sAb, long long sAh,
    const float* B, long long ldb, long long sBb, long long sBh, int transB,
    float* C, long long ldc, long long sCb, long long sCh,
    const float* bias, const float* resid, long long ldr, long long sRb, long long sRh,
    int M, int N, int K, int nb, int Hh, float alpha, int gelu) {
  GemmP p;
  p.A = A; p.B = B; p.C = C; p.bias = bias; p.resid = resid;
  p.lda = lda; p.ldb = ldb; p.ldc = ldc; p.ldr = ldr;
  p.sAb = sAb; p.sAh = sAh; p.sBb = sBb; p.sBh = sBh;
  p.sCb = sCb; p.sCh = sCh; p.sRb = sRb; p.sRh = sRh;
  p.M = M; p.N = N; p.K = K; p.Hh = Hh; p.transB = transB; p.gelu = gelu;
  p.alpha = alpha;
  dim3 grid((N + 127) / 128, (M + 127) / 128, nb * Hh);
  gemm_wmma_kernel<<<grid, 256, 0, st>>>(p);
}

extern "C" void kernel_launch(void* const* d_in, const int* in_sizes, int n_in,
                              void* d_out, int out_size, void* d_ws, size_t ws_size,
                              hipStream_t stream) {
  const float* x         = (const float*)d_in[0];
  const float* patch_w   = (const float*)d_in[1];
  const float* patch_b   = (const float*)d_in[2];
  const float* cls_token = (const float*)d_in[3];
  const float* pos_embed = (const float*)d_in[4];
  const float* sel_up_w  = (const float*)d_in[5];
  const float* sel_up_b  = (const float*)d_in[6];
  const float* sel_in_w  = (const float*)d_in[7];
  const float* sel_in_b  = (const float*)d_in[8];
  const float* sel_out_w = (const float*)d_in[9];
  const float* sel_out_b = (const float*)d_in[10];
  const float* sel_ln_s  = (const float*)d_in[11];
  const float* sel_ln_b  = (const float*)d_in[12];
  const float* sel_o1_w  = (const float*)d_in[13];
  const float* sel_o1_b  = (const float*)d_in[14];
  const float* sel_o2_w  = (const float*)d_in[15];
  const float* sel_o2_b  = (const float*)d_in[16];
  const float* blk_ln1_s = (const float*)d_in[17];
  const float* blk_ln1_b = (const float*)d_in[18];
  const float* blk_qkv_w = (const float*)d_in[19];
  const float* blk_qkv_b = (const float*)d_in[20];
  const float* blk_proj_w= (const float*)d_in[21];
  const float* blk_proj_b= (const float*)d_in[22];
  const float* blk_ln2_s = (const float*)d_in[23];
  const float* blk_ln2_b = (const float*)d_in[24];
  const float* blk_fc1_w = (const float*)d_in[25];
  const float* blk_fc1_b = (const float*)d_in[26];
  const float* blk_fc2_w = (const float*)d_in[27];
  const float* blk_fc2_b = (const float*)d_in[28];
  const float* norm_s    = (const float*)d_in[29];
  const float* norm_b    = (const float*)d_in[30];
  const float* head_w    = (const float*)d_in[31];
  const float* head_b    = (const float*)d_in[32];

  float* out = (float*)d_out;
  float* ws  = (float*)d_ws;

  // workspace layout (floats)
  const long long oX   = 0;                       // 6304*768
  const long long oX2  = 4841472LL;               // 6304*768
  const long long oY   = 9682944LL;               // 6304*1536
  const long long oH1  = 19365888LL;              // 6272*1536 (+pad)
  const long long oQKV = 29048832LL;              // 6304*4608 (multi-use)
  const long long oSC  = 58097664LL;              // 32*12*197*197
  const long long oG   = 72998016LL;              // 6304*768
  const long long oMK  = 77839488LL;              // 6272
  const long long need = 77845760LL;
  if (ws_size < (size_t)need * sizeof(float)) return;

  float* X   = ws + oX;
  float* X2  = ws + oX2;
  float* Y   = ws + oY;
  float* H1  = ws + oH1;
  float* QKV = ws + oQKV;
  float* SC  = ws + oSC;
  float* G   = ws + oG;
  float* MK  = ws + oMK;

  // ---- patch embed ----
  {
    long long tot = 6272LL * 768;
    im2col_kernel<<<(int)((tot + 255) / 256), 256, 0, stream>>>(x, QKV);
  }
  launch_gemm(stream, QKV, 768, 0, 0, patch_w, 768, 0, 0, 0,
              G, 768, 0, 0, patch_b, nullptr, 0, 0, 0,
              6272, 768, 768, 1, 1, 1.0f, 0);
  {
    long long tot = 32LL * 197 * 768;
    assemble_kernel<<<(int)((tot + 255) / 256), 256, 0, stream>>>(G, cls_token, pos_embed, X);
  }

  // ---- selector ----
  launch_gemm(stream, X + 768, 768, 197LL * 768, 0, sel_up_w, 1536, 0, 0, 0,
              H1, 1536, 196LL * 1536, 0, sel_up_b, nullptr, 0, 0, 0,
              196, 1536, 768, 32, 1, 1.0f, 0);
  launch_gemm(stream, H1, 1536, 0, 0, sel_in_w, 4608, 0, 0, 0,
              QKV, 4608, 0, 0, sel_in_b, nullptr, 0, 0, 0,
              6272, 4608, 1536, 1, 1, 1.0f, 0);
  // scores = (Q Kt) * 1/sqrt(192)   [32 x 8 heads]
  launch_gemm(stream, QKV, 4608, 196LL * 4608, 192,
              QKV + 1536, 4608, 196LL * 4608, 192, 1,
              SC, 196, 8LL * 196 * 196, 196LL * 196,
              nullptr, nullptr, 0, 0, 0,
              196, 196, 192, 32, 8, 1.0f / sqrtf(192.0f), 0);
  softmax_kernel<<<32 * 8 * 196, 256, 0, stream>>>(SC, 196);
  // att @ V -> Y
  launch_gemm(stream, SC, 196, 8LL * 196 * 196, 196LL * 196,
              QKV + 3072, 4608, 196LL * 4608, 192, 0,
              Y, 1536, 196LL * 1536, 192,
              nullptr, nullptr, 0, 0, 0,
              196, 192, 196, 32, 8, 1.0f, 0);
  // out proj + residual (H2 reuses QKV region)
  float* H2 = QKV;
  launch_gemm(stream, Y, 1536, 0, 0, sel_out_w, 1536, 0, 0, 0,
              H2, 1536, 0, 0, sel_out_b, H1, 1536, 0, 0,
              6272, 1536, 1536, 1, 1, 1.0f, 0);
  ln_kernel<<<6272, 256, 0, stream>>>(H2, H1, sel_ln_s, sel_ln_b, 1536, 1536, 1536, 1e-5f);
  launch_gemm(stream, H1, 1536, 0, 0, sel_o1_w, 768, 0, 0, 0,
              G, 768, 0, 0, sel_o1_b, nullptr, 0, 0, 0,
              6272, 768, 1536, 1, 1, 1.0f, 1);
  selmask_kernel<<<6272, 256, 0, stream>>>(G, sel_o2_w, sel_o2_b, MK, out + 32000);
  {
    long long tot = 6272LL * 768;
    applymask_kernel<<<(int)((tot + 255) / 256), 256, 0, stream>>>(X, MK);
  }

  // ---- 12 transformer blocks ----
  for (int l = 0; l < 12; l++) {
    const float* l1s = blk_ln1_s + (long long)l * 768;
    const float* l1b = blk_ln1_b + (long long)l * 768;
    const float* qw  = blk_qkv_w + (long long)l * 768 * 2304;
    const float* qb  = blk_qkv_b + (long long)l * 2304;
    const float* pw  = blk_proj_w + (long long)l * 768 * 768;
    const float* pb  = blk_proj_b + (long long)l * 768;
    const float* l2s = blk_ln2_s + (long long)l * 768;
    const float* l2b = blk_ln2_b + (long long)l * 768;
    const float* f1w = blk_fc1_w + (long long)l * 768 * 3072;
    const float* f1b = blk_fc1_b + (long long)l * 3072;
    const float* f2w = blk_fc2_w + (long long)l * 3072 * 768;
    const float* f2b = blk_fc2_b + (long long)l * 768;

    ln_kernel<<<6304, 256, 0, stream>>>(X, Y, l1s, l1b, 768, 768, 768, 1e-6f);
    launch_gemm(stream, Y, 768, 0, 0, qw, 2304, 0, 0, 0,
                QKV, 2304, 0, 0, qb, nullptr, 0, 0, 0,
                6304, 2304, 768, 1, 1, 1.0f, 0);
    launch_gemm(stream, QKV, 2304, 197LL * 2304, 64,
                QKV + 768, 2304, 197LL * 2304, 64, 1,
                SC, 197, 12LL * 197 * 197, 197LL * 197,
                nullptr, nullptr, 0, 0, 0,
                197, 197, 64, 32, 12, 0.125f, 0);
    softmax_kernel<<<32 * 12 * 197, 256, 0, stream>>>(SC, 197);
    launch_gemm(stream, SC, 197, 12LL * 197 * 197, 197LL * 197,
                QKV + 1536, 2304, 197LL * 2304, 64, 0,
                Y, 768, 197LL * 768, 64,
                nullptr, nullptr, 0, 0, 0,
                197, 64, 197, 32, 12, 1.0f, 0);
    launch_gemm(stream, Y, 768, 0, 0, pw, 768, 0, 0, 0,
                X2, 768, 0, 0, pb, X, 768, 0, 0,
                6304, 768, 768, 1, 1, 1.0f, 0);
    ln_kernel<<<6304, 256, 0, stream>>>(X2, Y, l2s, l2b, 768, 768, 768, 1e-6f);
    launch_gemm(stream, Y, 768, 0, 0, f1w, 3072, 0, 0, 0,
                QKV, 3072, 0, 0, f1b, nullptr, 0, 0, 0,
                6304, 3072, 768, 1, 1, 1.0f, 1);
    launch_gemm(stream, QKV, 3072, 0, 0, f2w, 768, 0, 0, 0,
                X, 768, 0, 0, f2b, X2, 768, 0, 0,
                6304, 768, 3072, 1, 1, 1.0f, 0);
  }

  // ---- final LN on cls rows only + head ----
  ln_kernel<<<32, 256, 0, stream>>>(X, X2, norm_s, norm_b, 768, 197LL * 768, 768, 1e-6f);
  launch_gemm(stream, X2, 768, 0, 0, head_w, 1000, 0, 0, 0,
              out, 1000, 0, 0, head_b, nullptr, 0, 0, 0,
              32, 1000, 768, 1, 1, 1.0f, 0);
}
